// MultiHeadAttention_83236466197181
// MI455X (gfx1250) — compile-verified, matrix operand order checked by reading, not run on hardware
//
#include <hip/hip_runtime.h>
#include <hip/hip_bf16.h>
#include <math.h>

typedef _Float16 f16;
typedef __attribute__((ext_vector_type(16))) _Float16 v16h;
typedef __attribute__((ext_vector_type(8)))  _Float16 v8h;
typedef __attribute__((ext_vector_type(4)))  _Float16 v4h;
typedef __attribute__((ext_vector_type(8)))  float    v8f;

#define BATCH 2
#define SEQ   4096
#define DMODEL 512
#define NHEAD 8
#define DHEAD 64
#define MROWS (BATCH * SEQ)   // 8192

// ---------------------------------------------------------------------------
// WMMA helper: D = A(16x32 f16) * B(32x16 f16) + C(16x16 f32)
// ---------------------------------------------------------------------------
__device__ __forceinline__ v8f wmma_f16(v16h a, v16h b, v8f c) {
    return __builtin_amdgcn_wmma_f32_16x16x32_f16(
        /*neg_a=*/false, a, /*neg_b=*/false, b,
        /*c_mod=*/(short)0, c, /*reuse_a=*/false, /*reuse_b=*/false);
}

// Load a 16x32 f16 fragment (A-layout; also valid as B fragment when the
// "B source" is stored transposed row-major, i.e. row n contiguous over k).
// base points at (row0, k0); ld = leading dimension in halves.
// Lane l: row = l&15 ; halves are k = kb..kb+7 and 16+kb..16+kb+7, kb=(l>>4)*8.
__device__ __forceinline__ v16h load_frag(const f16* __restrict__ base, int ld, int lane) {
    const f16* p = base + (size_t)(lane & 15) * ld + ((lane >> 4) << 3);
    v8h lo = *(const v8h*)(p);
    v8h hi = *(const v8h*)(p + 16);
    v16h r;
#pragma unroll
    for (int i = 0; i < 8; ++i) { r[i] = lo[i]; r[i + 8] = hi[i]; }
    return r;
}

// Same pattern but from LDS.
__device__ __forceinline__ v16h load_frag_lds(const f16* base, int ld, int lane) {
    const f16* p = base + (lane & 15) * ld + ((lane >> 4) << 3);
    v8h lo = *(const v8h*)(p);
    v8h hi = *(const v8h*)(p + 16);
    v16h r;
#pragma unroll
    for (int i = 0; i < 8; ++i) { r[i] = lo[i]; r[i + 8] = hi[i]; }
    return r;
}

// ---------------------------------------------------------------------------
// fp32 -> fp16 conversion (4 elements / thread, vectorized)
// ---------------------------------------------------------------------------
__global__ void cvt_f32_f16_kernel(const float* __restrict__ src,
                                   f16* __restrict__ dst, int n) {
    int i = (blockIdx.x * blockDim.x + threadIdx.x) * 4;
    if (i + 3 < n) {
        float4 v = *(const float4*)(src + i);
        v4h h = { (f16)v.x, (f16)v.y, (f16)v.z, (f16)v.w };
        *(v4h*)(dst + i) = h;
    } else {
        for (; i < n; ++i) dst[i] = (f16)src[i];
    }
}

// ---------------------------------------------------------------------------
// Projection GEMM: out[m,n] = sum_k in[m,k] * w[n,k] + bias[n]
// Register double-buffered K-loop: loads for step k+32 are issued BEFORE the
// WMMAs of step k, so the per-WMMA wait is loadcnt<=N, not 0.
// vt_mode==0: write f16 [B,H,S,Dh]   (Q, K)
// vt_mode==1: write f16 [B,H,Dh,S]   (V transposed, for contiguous P@V B-frags)
// Block: 256 threads = 8 waves; wave computes 16(M) x 64(N); WG = 128 x 64.
// ---------------------------------------------------------------------------
__global__ __launch_bounds__(256) void qkv_proj_kernel(
    const f16* __restrict__ xh, const f16* __restrict__ wh,
    const float* __restrict__ bias, f16* __restrict__ out, int vt_mode)
{
    const int lane = threadIdx.x & 31;
    const int wave = threadIdx.x >> 5;
    const int row0 = blockIdx.x * 128 + wave * 16;
    const int col0 = blockIdx.y * 64;

    v8f acc[4] = {};
    v16h a_cur = load_frag(xh + (size_t)row0 * DMODEL, DMODEL, lane);
    v16h b_cur[4];
#pragma unroll
    for (int j = 0; j < 4; ++j)
        b_cur[j] = load_frag(wh + (size_t)(col0 + j * 16) * DMODEL, DMODEL, lane);

#pragma unroll 2
    for (int k0 = 0; k0 < DMODEL; k0 += 32) {
        const int kn = (k0 + 32) & (DMODEL - 1);   // wraps to 0 on last iter
        v16h a_nxt = load_frag(xh + (size_t)row0 * DMODEL + kn, DMODEL, lane);
        v16h b_nxt[4];
#pragma unroll
        for (int j = 0; j < 4; ++j)
            b_nxt[j] = load_frag(wh + (size_t)(col0 + j * 16) * DMODEL + kn, DMODEL, lane);
#pragma unroll
        for (int j = 0; j < 4; ++j)
            acc[j] = wmma_f16(a_cur, b_cur[j], acc[j]);
        a_cur = a_nxt;
#pragma unroll
        for (int j = 0; j < 4; ++j) b_cur[j] = b_nxt[j];
    }

    const int rbase = row0 + ((lane >> 4) << 3);
#pragma unroll
    for (int j = 0; j < 4; ++j) {
        const int n  = col0 + j * 16 + (lane & 15);
        const float bn = bias[n];
        const int h  = n >> 6;
        const int dh = n & 63;
#pragma unroll
        for (int r = 0; r < 8; ++r) {
            const int gr = rbase + r;        // 0..8191
            const int b  = gr >> 12;
            const int s  = gr & (SEQ - 1);
            const f16 v = (f16)(acc[j][r] + bn);
            if (vt_mode)
                out[(((size_t)(b * NHEAD + h)) * DHEAD + dh) * SEQ + s] = v;
            else
                out[(((size_t)(b * NHEAD + h)) * SEQ + s) * DHEAD + dh] = v;
        }
    }
}

// ---------------------------------------------------------------------------
// Flash attention. One wave = 16 query rows; block = 4 waves = 64 rows.
// Grid = B*H*(S/64) = 1024. K/V tiles streamed from L2 (1 MB/head << 192 MB).
// Load schedule per tile: all K frags, then all V frags (in-order completion
// means QK^T only waits for K; V latency hides behind softmax + LDS), plus
// global_prefetch of the next K tile.
// ---------------------------------------------------------------------------
__global__ __launch_bounds__(128) void flash_attn_kernel(
    const f16* __restrict__ Q,   // [B,H,S,64]
    const f16* __restrict__ K,   // [B,H,S,64]
    const f16* __restrict__ Vt,  // [B,H,64,S]
    f16* __restrict__ O)         // [B,S,512] heads merged, f16
{
    __shared__ __align__(16) f16 lds_p[4][16 * 64];

    const int lane = threadIdx.x & 31;
    const int wave = threadIdx.x >> 5;
    const int bh   = blockIdx.x >> 6;       // (b*8 + h)
    const int rt   = blockIdx.x & 63;       // row tile within sequence
    const int b    = bh >> 3;
    const int h    = bh & 7;
    const int q0   = rt * 64 + wave * 16;   // first query row of this wave

    const f16* Qp = Q  + ((size_t)bh * SEQ + q0) * DHEAD;
    const f16* Kp = K  + (size_t)bh * SEQ * DHEAD;
    const f16* Vp = Vt + (size_t)bh * DHEAD * SEQ;

    const v16h aq0 = load_frag(Qp + 0,  DHEAD, lane);
    const v16h aq1 = load_frag(Qp + 32, DHEAD, lane);

    float mrow[8], lrow[8];
    v8f o[4] = {};
#pragma unroll
    for (int r = 0; r < 8; ++r) { mrow[r] = -1e30f; lrow[r] = 0.f; }

    f16* pbuf = lds_p[wave];
    const float scale = 0.125f;   // 1/sqrt(64)

    for (int t0 = 0; t0 < SEQ; t0 += 64) {
        // ---- issue ALL K fragment loads first (one clause) ----
        v16h bk0[4], bk1[4];
#pragma unroll
        for (int nt = 0; nt < 4; ++nt) {
            bk0[nt] = load_frag(Kp + (size_t)(t0 + nt * 16) * DHEAD + 0,  DHEAD, lane);
            bk1[nt] = load_frag(Kp + (size_t)(t0 + nt * 16) * DHEAD + 32, DHEAD, lane);
        }
        // ---- issue ALL V fragment loads next; consumed only after softmax ----
        v16h bv0[4], bv1[4];
#pragma unroll
        for (int j = 0; j < 4; ++j) {
            bv0[j] = load_frag(Vp + (size_t)(j * 16) * SEQ + t0 + 0,  SEQ, lane);
            bv1[j] = load_frag(Vp + (size_t)(j * 16) * SEQ + t0 + 32, SEQ, lane);
        }
        // ---- prefetch next K tile into cache (global_prefetch_b8) ----
        {
            const int tn = (t0 + 64) & (SEQ - 1);
            __builtin_prefetch(Kp + (size_t)(tn +  0 + (lane & 15)) * DHEAD, 0, 1);
            __builtin_prefetch(Kp + (size_t)(tn + 16 + (lane & 15)) * DHEAD, 0, 1);
            __builtin_prefetch(Kp + (size_t)(tn + 32 + (lane & 15)) * DHEAD, 0, 1);
            __builtin_prefetch(Kp + (size_t)(tn + 48 + (lane & 15)) * DHEAD, 0, 1);
        }

        // ---- S = Q @ K^T ----
        v8f sacc[4];
#pragma unroll
        for (int nt = 0; nt < 4; ++nt) {
            v8f c = {};
            c = wmma_f16(aq0, bk0[nt], c);
            c = wmma_f16(aq1, bk1[nt], c);
            sacc[nt] = c;
        }

        // ---- online softmax; lane owns rows (lane>>4)*8+r, col lane&15 ----
        float tmax[8];
#pragma unroll
        for (int r = 0; r < 8; ++r) {
            float v = sacc[0][r];
            v = fmaxf(v, sacc[1][r]); v = fmaxf(v, sacc[2][r]); v = fmaxf(v, sacc[3][r]);
            tmax[r] = v;
        }
        for (int mk = 1; mk < 16; mk <<= 1) {
#pragma unroll
            for (int r = 0; r < 8; ++r)
                tmax[r] = fmaxf(tmax[r], __shfl_xor(tmax[r], mk, 32));
        }
        float alpha[8];
#pragma unroll
        for (int r = 0; r < 8; ++r) {
            float mnew = fmaxf(mrow[r], tmax[r] * scale);
            alpha[r] = __expf(mrow[r] - mnew);
            mrow[r] = mnew;
        }
        float rsum[8];
#pragma unroll
        for (int r = 0; r < 8; ++r) rsum[r] = 0.f;
#pragma unroll
        for (int nt = 0; nt < 4; ++nt) {
#pragma unroll
            for (int r = 0; r < 8; ++r) {
                float p = __expf(sacc[nt][r] * scale - mrow[r]);
                rsum[r] += p;
                sacc[nt][r] = p;
            }
        }
        for (int mk = 1; mk < 16; mk <<= 1) {
#pragma unroll
            for (int r = 0; r < 8; ++r)
                rsum[r] += __shfl_xor(rsum[r], mk, 32);
        }
#pragma unroll
        for (int r = 0; r < 8; ++r) lrow[r] = lrow[r] * alpha[r] + rsum[r];
#pragma unroll
        for (int j = 0; j < 4; ++j)
#pragma unroll
            for (int r = 0; r < 8; ++r)
                o[j][r] *= alpha[r];

        // ---- P: C-layout -> A-layout via LDS ----
        __syncthreads();  // prior-iteration P reads done before overwrite
        const int rr0 = (lane >> 4) << 3;
#pragma unroll
        for (int nt = 0; nt < 4; ++nt)
#pragma unroll
            for (int r = 0; r < 8; ++r)
                pbuf[(rr0 + r) * 64 + nt * 16 + (lane & 15)] = (f16)sacc[nt][r];
        __syncthreads();

        const v16h ap0 = load_frag_lds(pbuf + 0,  64, lane);
        const v16h ap1 = load_frag_lds(pbuf + 32, 64, lane);

        // ---- O += P @ V  (V frags already in registers) ----
#pragma unroll
        for (int j = 0; j < 4; ++j) {
            o[j] = wmma_f16(ap0, bv0[j], o[j]);
            o[j] = wmma_f16(ap1, bv1[j], o[j]);
        }
    }

    // ---- normalize and store merged-head f16 output ----
    float rinv[8];
#pragma unroll
    for (int r = 0; r < 8; ++r) rinv[r] = 1.0f / lrow[r];
    const int rr0 = (lane >> 4) << 3;
#pragma unroll
    for (int j = 0; j < 4; ++j) {
        const int dh = j * 16 + (lane & 15);
#pragma unroll
        for (int r = 0; r < 8; ++r) {
            const int s = q0 + rr0 + r;
            O[((size_t)b * SEQ + s) * DMODEL + h * DHEAD + dh] = (f16)(o[j][r] * rinv[r]);
        }
    }
}

// ---------------------------------------------------------------------------
// Output projection: out[m,n] = sum_k Oh[m,k]*wo[n,k] + bo[n]   (f32 store)
// Same double-buffered pipeline as qkv_proj.
// ---------------------------------------------------------------------------
__global__ __launch_bounds__(256) void out_proj_kernel(
    const f16* __restrict__ oh, const f16* __restrict__ wh,
    const float* __restrict__ bias, float* __restrict__ out)
{
    const int lane = threadIdx.x & 31;
    const int wave = threadIdx.x >> 5;
    const int row0 = blockIdx.x * 128 + wave * 16;
    const int col0 = blockIdx.y * 64;

    v8f acc[4] = {};
    v16h a_cur = load_frag(oh + (size_t)row0 * DMODEL, DMODEL, lane);
    v16h b_cur[4];
#pragma unroll
    for (int j = 0; j < 4; ++j)
        b_cur[j] = load_frag(wh + (size_t)(col0 + j * 16) * DMODEL, DMODEL, lane);

#pragma unroll 2
    for (int k0 = 0; k0 < DMODEL; k0 += 32) {
        const int kn = (k0 + 32) & (DMODEL - 1);
        v16h a_nxt = load_frag(oh + (size_t)row0 * DMODEL + kn, DMODEL, lane);
        v16h b_nxt[4];
#pragma unroll
        for (int j = 0; j < 4; ++j)
            b_nxt[j] = load_frag(wh + (size_t)(col0 + j * 16) * DMODEL + kn, DMODEL, lane);
#pragma unroll
        for (int j = 0; j < 4; ++j)
            acc[j] = wmma_f16(a_cur, b_cur[j], acc[j]);
        a_cur = a_nxt;
#pragma unroll
        for (int j = 0; j < 4; ++j) b_cur[j] = b_nxt[j];
    }

    const int rbase = row0 + ((lane >> 4) << 3);
#pragma unroll
    for (int j = 0; j < 4; ++j) {
        const int n = col0 + j * 16 + (lane & 15);
        const float bn = bias[n];
#pragma unroll
        for (int r = 0; r < 8; ++r)
            out[(size_t)(rbase + r) * DMODEL + n] = acc[j][r] + bn;
    }
}

// ---------------------------------------------------------------------------
extern "C" void kernel_launch(void* const* d_in, const int* in_sizes, int n_in,
                              void* d_out, int out_size, void* d_ws, size_t ws_size,
                              hipStream_t stream) {
    const float* x  = (const float*)d_in[0];
    const float* wq = (const float*)d_in[1];
    const float* bq = (const float*)d_in[2];
    const float* wk = (const float*)d_in[3];
    const float* bk = (const float*)d_in[4];
    const float* wv = (const float*)d_in[5];
    const float* bv = (const float*)d_in[6];
    const float* wo = (const float*)d_in[7];
    const float* bo = (const float*)d_in[8];
    float* out = (float*)d_out;

    const size_t nX = (size_t)MROWS * DMODEL;       // 4,194,304
    const size_t nW = (size_t)DMODEL * DMODEL;      //   262,144

    char* p = (char*)d_ws;
    f16* xh  = (f16*)p; p += nX * sizeof(f16);
    f16* wqh = (f16*)p; p += nW * sizeof(f16);
    f16* wkh = (f16*)p; p += nW * sizeof(f16);
    f16* wvh = (f16*)p; p += nW * sizeof(f16);
    f16* woh = (f16*)p; p += nW * sizeof(f16);
    f16* Qb  = (f16*)p; p += nX * sizeof(f16);
    f16* Kb  = (f16*)p; p += nX * sizeof(f16);
    f16* Vtb = (f16*)p; p += nX * sizeof(f16);
    f16* Oh  = (f16*)p; p += nX * sizeof(f16);

    // 1) fp32 -> fp16 conversions (4 elems/thread)
    cvt_f32_f16_kernel<<<(unsigned)((nX / 4 + 255) / 256), 256, 0, stream>>>(x,  xh,  (int)nX);
    cvt_f32_f16_kernel<<<(unsigned)((nW / 4 + 255) / 256), 256, 0, stream>>>(wq, wqh, (int)nW);
    cvt_f32_f16_kernel<<<(unsigned)((nW / 4 + 255) / 256), 256, 0, stream>>>(wk, wkh, (int)nW);
    cvt_f32_f16_kernel<<<(unsigned)((nW / 4 + 255) / 256), 256, 0, stream>>>(wv, wvh, (int)nW);
    cvt_f32_f16_kernel<<<(unsigned)((nW / 4 + 255) / 256), 256, 0, stream>>>(wo, woh, (int)nW);

    // 2) QKV projections (V written transposed per head)
    dim3 pg(MROWS / 128, DMODEL / 64);
    qkv_proj_kernel<<<pg, 256, 0, stream>>>(xh, wqh, bq, Qb,  0);
    qkv_proj_kernel<<<pg, 256, 0, stream>>>(xh, wkh, bk, Kb,  0);
    qkv_proj_kernel<<<pg, 256, 0, stream>>>(xh, wvh, bv, Vtb, 1);

    // 3) fused flash attention -> merged-head f16
    flash_attn_kernel<<<BATCH * NHEAD * (SEQ / 64), 128, 0, stream>>>(Qb, Kb, Vtb, Oh);

    // 4) output projection -> fp32
    out_proj_kernel<<<pg, 256, 0, stream>>>(Oh, woh, bo, out);
}
